// SubgraphEmbedder_70411693851276
// MI455X (gfx1250) — matrix operand order repaired and checked by reading, not compiled, not run on hardware
//
#include <hip/hip_runtime.h>
#include <hip/hip_bf16.h>

typedef __bf16 bf16_t;
typedef __attribute__((ext_vector_type(16))) __bf16 v16bf;
typedef __attribute__((ext_vector_type(8)))  __bf16 v8bf;
typedef __attribute__((ext_vector_type(8)))  float  v8f;

__device__ __forceinline__ float sigmoidf_(float x) { return 1.0f / (1.0f + __expf(-x)); }

// ---- A-fragment loaders (16x32 bf16 fragment, ISA 7.12.2 layout) ------------
// lanes 0-15 hold K=kb+{0..7,16..23}; lanes 16-31 hold K=kb+{8..15,24..31}
__device__ __forceinline__ v16bf load_afrag(const float* __restrict__ arow, int kb, int half)
{
    const float* ap = arow + kb + half * 8;
    float4 f0 = *(const float4*)(ap);
    float4 f1 = *(const float4*)(ap + 4);
    float4 f2 = *(const float4*)(ap + 16);
    float4 f3 = *(const float4*)(ap + 20);
    v16bf a;
    a[0]  = (bf16_t)f0.x; a[1]  = (bf16_t)f0.y; a[2]  = (bf16_t)f0.z; a[3]  = (bf16_t)f0.w;
    a[4]  = (bf16_t)f1.x; a[5]  = (bf16_t)f1.y; a[6]  = (bf16_t)f1.z; a[7]  = (bf16_t)f1.w;
    a[8]  = (bf16_t)f2.x; a[9]  = (bf16_t)f2.y; a[10] = (bf16_t)f2.z; a[11] = (bf16_t)f2.w;
    a[12] = (bf16_t)f3.x; a[13] = (bf16_t)f3.y; a[14] = (bf16_t)f3.z; a[15] = (bf16_t)f3.w;
    return a;
}
__device__ __forceinline__ v16bf load_afrag(const bf16_t* __restrict__ arow, int kb, int half)
{
    const bf16_t* ap = arow + kb + half * 8;
    v8bf lo = *(const v8bf*)(ap);
    v8bf hi = *(const v8bf*)(ap + 16);
    v16bf a;
    #pragma unroll
    for (int r = 0; r < 8; ++r) { a[r] = lo[r]; a[8 + r] = hi[r]; }
    return a;
}
__device__ __forceinline__ void store_out(float*  p, float v) { *p = v; }
__device__ __forceinline__ void store_out(bf16_t* p, float v) { *p = (bf16_t)v; }

// ---------------------------------------------------------------------------
// WMMA GEMM: C[M x Nout] = act(A[M x Kpad] @ W + bias). W pre-transposed as
// bf16 WT[Nout][Kpad] (zero padded past true K). A f32 or bf16; C f32 or bf16.
// One wave = one 16x16 tile via v_wmma_f32_16x16x32_bf16.
// Block = 256 threads = 8 waves = 32 rows x 64 cols. ACT: 0 none, 1 relu, 2 leaky(0.1)
// ---------------------------------------------------------------------------
template<int ACT, typename AT, typename OT>
__global__ void __launch_bounds__(256)
wmma_gemm(const AT* __restrict__ A, const bf16_t* __restrict__ WT,
          const float* __restrict__ bias, OT* __restrict__ C,
          int M, int Kpad, int Nout)
{
    const int lane = threadIdx.x & 31;
    const int wave = threadIdx.x >> 5;
    const int half = lane >> 4;
    const int l16  = lane & 15;

    const int mtile   = blockIdx.x * 2 + (wave >> 2);
    const int rowbase = mtile * 16;
    const int ncol    = blockIdx.y * 64 + (wave & 3) * 16 + l16;

    const int arow_idx = min(rowbase + l16, M - 1);     // clamp for edge tile
    const AT*     arow = A  + (size_t)arow_idx * Kpad;
    const bf16_t* brow = WT + (size_t)ncol     * Kpad;

    v8f acc = {};
    const int ksteps = Kpad >> 5;
    for (int ks = 0; ks < ksteps; ++ks) {
        const int kb = ks << 5;
        if (ks + 1 < ksteps) __builtin_prefetch(arow + kb + 32, 0, 1); // global_prefetch_b8
        v16bf a = load_afrag(arow, kb, half);
        // B 32x16 bf16 fragment: lane half 0 -> K=kb+{0..7,16..23}, half 1 -> K=kb+{8..15,24..31},
        // column = l16 -> two 16B runs in transposed WT.
        v8bf blo = *(const v8bf*)(brow + kb + half * 8);
        v8bf bhi = *(const v8bf*)(brow + kb + half * 8 + 16);
        v16bf b;
        #pragma unroll
        for (int r = 0; r < 8; ++r) { b[r] = blo[r]; b[8 + r] = bhi[r]; }
        acc = __builtin_amdgcn_wmma_f32_16x16x32_bf16(false, a, false, b, (short)0, acc,
                                                      false, false);
    }

    // C/D layout: reg r -> row M = rowbase + r + 8*half, col N = ncol
    const float bv = bias[ncol];
    #pragma unroll
    for (int r = 0; r < 8; ++r) {
        int mg = rowbase + r + half * 8;
        if (mg >= M) continue;
        float v = acc[r] + bv;
        if (ACT == 1) v = fmaxf(v, 0.0f);
        if (ACT == 2) v = v > 0.0f ? v : 0.1f * v;
        store_out(C + (size_t)mg * Nout + ncol, v);
    }
}

// Transpose + f32->bf16 convert weight W[K][Nout] -> WT[Nout][Kpad], zero pad K..Kpad
__global__ void wt_prep(const float* __restrict__ W, bf16_t* __restrict__ WT,
                        int K, int Nout, int Kpad)
{
    int t = blockIdx.x * blockDim.x + threadIdx.x;
    if (t >= Nout * Kpad) return;
    int n = t / Kpad, k = t % Kpad;
    WT[t] = (k < K) ? (bf16_t)W[(size_t)k * Nout + n] : (bf16_t)0.0f;
}

// Pad node features [n][feat] -> [n][fpad] with zeros
__global__ void pad_feat(const float* __restrict__ X, float* __restrict__ Xp,
                         int n, int feat, int fpad)
{
    int t = blockIdx.x * blockDim.x + threadIdx.x;
    if (t >= n * fpad) return;
    int node = t / fpad, c = t % fpad;
    Xp[t] = (c < feat) ? X[(size_t)node * feat + c] : 0.0f;
}

// Layer-invariant block aggregation: S_j[dst] += ew * E_j[src]  (computed ONCE per block)
// One wave per edge; lane handles channels {2l, 2l+1}.
__global__ void edge_scatter_block(const int* __restrict__ ei, const float* __restrict__ ew,
                                   const float* __restrict__ Ej, float* __restrict__ Sj,
                                   int nEdges)
{
    size_t t = (size_t)blockIdx.x * blockDim.x + threadIdx.x;
    int e = (int)(t >> 5);
    int lane = (int)(t & 31);
    if (e >= nEdges) return;
    int src = ei[e];
    int dst = ei[nEdges + e];
    if (src == dst) return;                 // reference zeroes self-loop weights
    float w = ew[e];
    int c = lane * 2;
    float2 v = *(const float2*)(Ej + (size_t)src * 64 + c);
    atomicAdd(&Sj[(size_t)dst * 64 + c],     v.x * w);
    atomicAdd(&Sj[(size_t)dst * 64 + c + 1], v.y * w);
}

// hbuf[node][j][c] = sigmoid(skip[layer][j]) * (E_j[node][c] + S_j[node][c])
__global__ void hbuf_init(const float* __restrict__ skip, const float* __restrict__ E,
                          const float* __restrict__ S, float* __restrict__ hbuf,
                          int nb, int layer, int nNodes)
{
    size_t t = (size_t)blockIdx.x * blockDim.x + threadIdx.x;
    size_t total = (size_t)nNodes * nb * 64;
    if (t >= total) return;
    int c = (int)(t & 63);
    int j = (int)((t >> 6) % nb);
    int node = (int)(t / ((size_t)nb * 64));
    float g = sigmoidf_(skip[layer * 4 + j]);
    size_t eoff = ((size_t)j * nNodes + node) * 64 + c;
    hbuf[t] = g * (E[eoff] + S[eoff]);
}

// Segment-sum pooling without atomics: batch is sorted, one block (320 thr) per graph;
// each block binary-searches its node range and reduces channel-wise.
__global__ void pool_seg(const int* __restrict__ batch, const float* __restrict__ E,
                         float* __restrict__ pooled, int nNodes)
{
    int g = blockIdx.x;
    int c = threadIdx.x;            // 0..319
    int lo = 0, hi = nNodes;
    while (lo < hi) { int mid = (lo + hi) >> 1; if (batch[mid] < g) lo = mid + 1; else hi = mid; }
    int start = lo;
    hi = nNodes;
    while (lo < hi) { int mid = (lo + hi) >> 1; if (batch[mid] <= g) lo = mid + 1; else hi = mid; }
    int end = lo;
    int j = c >> 6, cc = c & 63;
    const float* base = E + (size_t)j * nNodes * 64 + cc;
    float acc = 0.0f;
    for (int node = start; node < end; ++node) acc += base[(size_t)node * 64];
    pooled[(size_t)g * 320 + c] = acc;
}

extern "C" void kernel_launch(void* const* d_in, const int* in_sizes, int n_in,
                              void* d_out, int out_size, void* d_ws, size_t ws_size,
                              hipStream_t stream)
{
    const float* nf    = (const float*)d_in[0];
    const int*   ei    = (const int*)d_in[1];
    const float* ewt   = (const float*)d_in[2];
    const int*   batch = (const int*)d_in[3];
    const float* b_pre = (const float*)d_in[5];
    const float* skip  = (const float*)d_in[6];

    const int nNodes  = in_sizes[3];
    const int nEdges  = in_sizes[2];
    const int nGraphs = out_size / 64;

    size_t off = 0;
    auto ws = [&](size_t bytes) -> void* {
        void* p = (char*)d_ws + off;
        off += (bytes + 255) & ~(size_t)255;
        return p;
    };

    float*  nf_pad = (float*)ws((size_t)nNodes * 96 * 4);
    float*  E      = (float*)ws((size_t)nNodes * 5 * 64 * 4);  // E_j = E + j*nNodes*64
    float*  S      = (float*)ws((size_t)nNodes * 4 * 64 * 4);  // S_j = S + j*nNodes*64
    float*  hbuf   = (float*)ws((size_t)nNodes * 256 * 4);
    bf16_t* h1     = (bf16_t*)ws((size_t)nNodes * 64 * 2);
    float*  pooled = (float*)ws((size_t)nGraphs * 320 * 4);
    bf16_t* pp1    = (bf16_t*)ws((size_t)nGraphs * 64 * 2);
    bf16_t* pp2    = (bf16_t*)ws((size_t)nGraphs * 64 * 2);
    bf16_t* pp3    = (bf16_t*)ws((size_t)nGraphs * 256 * 2);

    // 13 weight matrices: {d_in idx, K, Nout, Kpad}
    struct WtDesc { int idx, K, N, Kpad; bf16_t* wt; };
    WtDesc wts[13] = {
        {4,  65,  64,  96, nullptr},                              // w_pre
        {7,  64,  64,  64, nullptr}, {9,  64, 64, 64, nullptr},   // w1_0, w2_0
        {11, 128, 64, 128, nullptr}, {13, 64, 64, 64, nullptr},   // w1_1, w2_1
        {15, 192, 64, 192, nullptr}, {17, 64, 64, 64, nullptr},   // w1_2, w2_2
        {19, 256, 64, 256, nullptr}, {21, 64, 64, 64, nullptr},   // w1_3, w2_3
        {23, 320, 64, 320, nullptr},                              // wp1
        {25, 64,  64,  64, nullptr},                              // wp2
        {27, 64, 256,  64, nullptr},                              // wp3
        {29, 256, 64, 256, nullptr},                              // wp4
    };
    for (int i = 0; i < 13; ++i)
        wts[i].wt = (bf16_t*)ws((size_t)wts[i].N * wts[i].Kpad * sizeof(bf16_t));
    for (int i = 0; i < 13; ++i) {
        int total = wts[i].N * wts[i].Kpad;
        wt_prep<<<(total + 255) / 256, 256, 0, stream>>>(
            (const float*)d_in[wts[i].idx], wts[i].wt, wts[i].K, wts[i].N, wts[i].Kpad);
    }

    { int total = nNodes * 96;
      pad_feat<<<(total + 255) / 256, 256, 0, stream>>>(nf, nf_pad, nNodes, 65, 96); }

    const dim3 gN((nNodes + 31) / 32, 1), gG((nGraphs + 31) / 32, 1), b256(256);

    // x0 = nf @ w_pre + b_pre (no activation)
    wmma_gemm<0, float, float><<<gN, b256, 0, stream>>>(
        nf_pad, wts[0].wt, b_pre, E, nNodes, 96, 64);

    const int w1idx[4] = {1, 3, 5, 7};
    const int w2idx[4] = {2, 4, 6, 8};
    const unsigned edgeBlocks = (unsigned)(((size_t)nEdges * 32 + 255) / 256);
    for (int i = 0; i < 4; ++i) {
        int nb = i + 1;
        float* Si = S + (size_t)i * nNodes * 64;
        // S_i = segment_sum(ew * E_i[src], dst)   -- computed once, reused by layers >= i
        (void)hipMemsetAsync(Si, 0, (size_t)nNodes * 64 * 4, stream);
        edge_scatter_block<<<edgeBlocks, b256, 0, stream>>>(
            ei, ewt, E + (size_t)i * nNodes * 64, Si, nEdges);
        { size_t total = (size_t)nNodes * nb * 64;
          hbuf_init<<<(unsigned)((total + 255) / 256), b256, 0, stream>>>(
              skip, E, S, hbuf, nb, i, nNodes); }
        // h1 = relu(hbuf @ w1_i + b1_i)  (bf16 out: consumed only by next GEMM)
        wmma_gemm<1, float, bf16_t><<<gN, b256, 0, stream>>>(
            hbuf, wts[w1idx[i]].wt, (const float*)d_in[8 + 4 * i], h1, nNodes, nb * 64, 64);
        // E_{i+1} = relu(h1 @ w2_i + b2_i)
        wmma_gemm<1, bf16_t, float><<<gN, b256, 0, stream>>>(
            h1, wts[w2idx[i]].wt, (const float*)d_in[10 + 4 * i],
            E + (size_t)(i + 1) * nNodes * 64, nNodes, 64, 64);
    }

    pool_seg<<<nGraphs, 320, 0, stream>>>(batch, E, pooled, nNodes);

    wmma_gemm<2, float,  bf16_t><<<gG, b256, 0, stream>>>(
        pooled, wts[9].wt, (const float*)d_in[24], pp1, nGraphs, 320, 64);
    wmma_gemm<1, bf16_t, bf16_t><<<gG, b256, 0, stream>>>(
        pp1, wts[10].wt, (const float*)d_in[26], pp2, nGraphs, 64, 64);
    wmma_gemm<1, bf16_t, bf16_t><<<dim3(gG.x, 4), b256, 0, stream>>>(
        pp2, wts[11].wt, (const float*)d_in[28], pp3, nGraphs, 64, 256);
    wmma_gemm<0, bf16_t, float><<<gG, b256, 0, stream>>>(
        pp3, wts[12].wt, (const float*)d_in[30], (float*)d_out, nGraphs, 256, 64);
}
